// GAFM_Item_35381940584972
// MI455X (gfx1250) — matrix operand order repaired and checked by reading, not compiled, not run on hardware
//
#include <hip/hip_runtime.h>
#include <math.h>

typedef __attribute__((ext_vector_type(16))) _Float16 v16h;
typedef __attribute__((ext_vector_type(8)))  _Float16 v8h;
typedef __attribute__((ext_vector_type(8)))  float    v8f;

#define KD   64      // embedding dim
#define TD   64      // attention hidden dim
#define FAN  20      // neighbors per node
#define GRP  16      // nodes per wave (== WMMA M)
#define WPB  4       // waves per block
#define BLK  (32*WPB)

union V16U { v16h v; v8h h[2]; };

// One wave handles GRP=16 nodes:
//   phase 1: gather 20 neighbor rows + target row (coalesced 256B rows, wave-
//            uniform indices -> scalar index loads), FM aggregator in f32.
//   phase 2: attention MLP: (tgt*fm)[16x64] @ Wa[64x64] via 8x
//            v_wmma_f32_16x16x32_f16 (f32 accum); fragments are built from
//            contiguous 16B LDS runs (ds_load_b128), Wa held transposed.
//            +ba, relu, @Wh via butterfly reduction; softmax over a size-1
//            axis == 1.0 exactly (kept live through the WMMA chain).
//   phase 3: combine atts*fm + tgt; hop0 writes agg rows, hop1 fuses the
//            user-embedding dot + sigmoid.
template<bool HOP1>
__global__ __launch_bounds__(BLK)
void gafm_hop_kernel(const int*   __restrict__ tgt_idx,
                     const int*   __restrict__ nbr_idx,
                     const float* __restrict__ nbr_src,   // entity emb (hop0) or agg0 (hop1)
                     const float* __restrict__ ent_emb,
                     const float* __restrict__ Wa,
                     const float* __restrict__ ba,
                     const float* __restrict__ Wh,
                     const float* __restrict__ bh,
                     float*       __restrict__ agg_out,   // hop0 only
                     const int*   __restrict__ u_idx,     // hop1 only
                     const float* __restrict__ users_emb, // hop1 only
                     float*       __restrict__ score_out) // hop1 only
{
    __shared__ _Float16 sWaT[TD * KD];       // f16 Wa TRANSPOSED: [t][k]
    __shared__ float    sWh[TD];
    __shared__ float    sBa[TD];
    __shared__ float    sTgt[WPB][GRP][KD];
    __shared__ float    sFm [WPB][GRP][KD];
    __shared__ _Float16 sTg [WPB][GRP][KD];  // f16(tgt*fm) = WMMA A operand
    __shared__ float    sAtt[WPB][GRP];

    const int tid = threadIdx.x;
    // transpose Wa (K,T) -> sWaT[t][k] so B fragments are contiguous in LDS
    for (int i = tid; i < KD * TD; i += BLK) {
        const int t = i >> 6;                 // dest row    (TD=64)
        const int k = i & 63;                 // dest column (KD=64)
        sWaT[i] = (_Float16)Wa[k * TD + t];
    }
    for (int i = tid; i < TD; i += BLK) { sWh[i] = Wh[i]; sBa[i] = ba[i]; }
    __syncthreads();

    const int wave = tid >> 5;
    const int lane = tid & 31;
    const int kk   = lane << 1;                       // this lane's two K slots
    const int nodeBase = (blockIdx.x * WPB + wave) * GRP;

    // ---- phase 1: gather + FM aggregator (f32) -----------------------------
    for (int m = 0; m < GRP; ++m) {
        const int node = nodeBase + m;                // wave-uniform
        const int ti = tgt_idx[node];
        const float2 tv = *(const float2*)(ent_emb + (size_t)ti * KD + kk);
        float s0 = 0.f, s1 = 0.f, q0 = 0.f, q1 = 0.f;
        for (int j = 0; j < FAN; ++j) {
            const int ni = nbr_idx[node * FAN + j];   // wave-uniform -> s_load
            const float2 e = *(const float2*)(nbr_src + (size_t)ni * KD + kk);
            s0 += e.x; s1 += e.y;
            q0 += e.x * e.x; q1 += e.y * e.y;
        }
        const float f0 = s0 * s0 - q0;
        const float f1 = s1 * s1 - q1;
        *(float2*)&sTgt[wave][m][kk] = tv;
        *(float2*)&sFm [wave][m][kk] = make_float2(f0, f1);
        sTg[wave][m][kk]     = (_Float16)(tv.x * f0);
        sTg[wave][m][kk + 1] = (_Float16)(tv.y * f1);
    }
    __syncthreads();

    // ---- phase 2: attention via WMMA --------------------------------------
    const int hlf  = lane >> 4;      // 0: lanes 0-15, 1: lanes 16-31
    const int mrow = lane & 15;      // A row (M) / C column (N) sub-index

    // ISA 16-bit A 16x32 layout: per-lane elements are two contiguous 8-half
    // runs: K = 8*hlf + [0..7] and K = 16 + 8*hlf + [0..7]  -> 2x ds_load_b128
    V16U a0, a1;
    a0.h[0] = *(const v8h*)&sTg[wave][mrow][     hlf * 8];
    a0.h[1] = *(const v8h*)&sTg[wave][mrow][16 + hlf * 8];
    a1.h[0] = *(const v8h*)&sTg[wave][mrow][32 + hlf * 8];
    a1.h[1] = *(const v8h*)&sTg[wave][mrow][48 + hlf * 8];

    float preg[8];
    #pragma unroll
    for (int r = 0; r < 8; ++r) preg[r] = 0.f;

    #pragma unroll
    for (int nt = 0; nt < 4; ++nt) {                 // four 16-wide N tiles
        const int bn = nt * 16 + mrow;               // output column
        // B 32x16 layout matching A: K = 8*hlf + [0..7] and K = 16 + 8*hlf + [0..7];
        // with sWaT[t][k] a fragment is two contiguous 16B runs per lane.
        const _Float16* wcol = &sWaT[bn * KD];
        V16U b0, b1;
        b0.h[0] = *(const v8h*)&wcol[     hlf * 8];
        b0.h[1] = *(const v8h*)&wcol[16 + hlf * 8];
        b1.h[0] = *(const v8h*)&wcol[32 + hlf * 8];
        b1.h[1] = *(const v8h*)&wcol[48 + hlf * 8];

        v8f acc = {0.f, 0.f, 0.f, 0.f, 0.f, 0.f, 0.f, 0.f};
        acc = __builtin_amdgcn_wmma_f32_16x16x32_f16(false, a0.v, false, b0.v,
                                                     (short)0, acc, false, false);
        acc = __builtin_amdgcn_wmma_f32_16x16x32_f16(false, a1.v, false, b1.v,
                                                     (short)0, acc, false, false);
        const float ban = sBa[bn];
        const float whn = sWh[bn];
        #pragma unroll
        for (int r = 0; r < 8; ++r) {                // M = r + 8*hlf
            float h = acc[r] + ban;
            h = fmaxf(h, 0.f);                       // relu
            preg[r] += h * whn;                      // partial of h @ Wh
        }
    }
    // reduce the column partials across the 16 lanes of each half-wave
    #pragma unroll
    for (int off = 1; off < 16; off <<= 1) {
        #pragma unroll
        for (int r = 0; r < 8; ++r)
            preg[r] += __shfl_xor(preg[r], off, 32);
    }
    if (mrow == 0) {
        const float bhv = bh[0];
        #pragma unroll
        for (int r = 0; r < 8; ++r) {
            const float z  = preg[r] + bhv;
            const float ez = expf(z - z);            // stable softmax numerator
            sAtt[wave][hlf * 8 + r] = ez / ez;       // size-1 softmax == 1.0
        }
    }
    __syncthreads();

    // ---- phase 3: residual combine (+ fused scoring on hop1) ---------------
    if (!HOP1) {
        for (int m = 0; m < GRP; ++m) {
            const float  a  = sAtt[wave][m];
            const float2 fm = *(const float2*)&sFm [wave][m][kk];
            const float2 tg = *(const float2*)&sTgt[wave][m][kk];
            float2 o; o.x = a * fm.x + tg.x; o.y = a * fm.y + tg.y;
            *(float2*)(agg_out + (size_t)(nodeBase + m) * KD + kk) = o;
        }
    } else {
        for (int m = 0; m < GRP; ++m) {
            const int node = nodeBase + m;
            const float  a  = sAtt[wave][m];
            const int uid   = u_idx[node];
            const float2 ue = *(const float2*)(users_emb + (size_t)uid * KD + kk);
            const float2 fm = *(const float2*)&sFm [wave][m][kk];
            const float2 tg = *(const float2*)&sTgt[wave][m][kk];
            const float g0 = a * fm.x + tg.x;
            const float g1 = a * fm.y + tg.y;
            float p = ue.x * g0 + ue.y * g1;
            #pragma unroll
            for (int off = 1; off < 32; off <<= 1)
                p += __shfl_xor(p, off, 32);
            if (lane == 0) score_out[node] = 1.f / (1.f + expf(-p));
        }
    }
}

extern "C" void kernel_launch(void* const* d_in, const int* in_sizes, int n_in,
                              void* d_out, int out_size, void* d_ws, size_t ws_size,
                              hipStream_t stream) {
    const int*   u        = (const int*)  d_in[0];
    const int*   tgt1     = (const int*)  d_in[1];
    const int*   nbr1_pos = (const int*)  d_in[2];
    const int*   tgt0     = (const int*)  d_in[3];
    const int*   nbr0     = (const int*)  d_in[4];
    const float* ent      = (const float*)d_in[5];
    const float* usr      = (const float*)d_in[6];
    const float* Wa       = (const float*)d_in[7];
    const float* ba       = (const float*)d_in[8];
    const float* Wh       = (const float*)d_in[9];
    const float* bh       = (const float*)d_in[10];
    float* out  = (float*)d_out;
    float* agg0 = (float*)d_ws;          // N0 * KD floats = 20.97 MB staging

    const int B  = 4096;
    const int N0 = B * FAN;              // 81920

    dim3 blk(BLK);
    dim3 g0(N0 / (WPB * GRP));           // 1280 blocks
    dim3 g1(B  / (WPB * GRP));           // 64 blocks

    hipLaunchKernelGGL((gafm_hop_kernel<false>), g0, blk, 0, stream,
                       tgt0, nbr0, ent, ent, Wa, ba, Wh, bh,
                       agg0, (const int*)nullptr, (const float*)nullptr,
                       (float*)nullptr);
    hipLaunchKernelGGL((gafm_hop_kernel<true>), g1, blk, 0, stream,
                       tgt1, nbr1_pos, agg0, ent, Wa, ba, Wh, bh,
                       (float*)nullptr, u, usr, out);
}
